// MOA_13254269075617
// MI455X (gfx1250) — matrix operand order repaired and checked by reading, not compiled, not run on hardware
//
#include <hip/hip_runtime.h>
#include <math.h>

// ---------------------------------------------------------------------------
// MI455X / gfx1250 implementation of the outlook-attention block.
// bf16 WMMA (v_wmma_f32_16x16x32_bf16) for all three GEMMs, f32 accumulate.
// K-loop software-pipelined by one step; sched_barrier(0) fences keep the
// next-step fragment loads grouped ahead of the current-step WMMAs so the
// backend waits on a partial loadcnt instead of serializing load->wmma.
// ---------------------------------------------------------------------------

typedef __bf16 bf16_t;
typedef bf16_t v4bf  __attribute__((ext_vector_type(4)));
typedef bf16_t v8bf  __attribute__((ext_vector_type(8)));
typedef bf16_t v16bf __attribute__((ext_vector_type(16)));
typedef float  v8fv  __attribute__((ext_vector_type(8)));
typedef float  v8f   __attribute__((ext_vector_type(8)));

#define BB    8
#define HWDIM 64
#define CDIM  256
#define NPIX  32768   // B*H*W
#define EPSBN 1e-5f

#if __has_builtin(__builtin_amdgcn_sched_barrier)
#define SCHED_FENCE() __builtin_amdgcn_sched_barrier(0)
#else
#define SCHED_FENCE()
#endif

// ---------------------------------------------------------------------------
// helpers
// ---------------------------------------------------------------------------
template <typename T> struct vec8;
template <> struct vec8<float>  { typedef v8fv type; };
template <> struct vec8<bf16_t> { typedef v8bf type; };

__device__ __forceinline__ float4 ld4(const float* p) { return *(const float4*)p; }
__device__ __forceinline__ float4 max4(float4 a, float4 b) {
    return make_float4(fmaxf(a.x,b.x), fmaxf(a.y,b.y), fmaxf(a.z,b.z), fmaxf(a.w,b.w));
}
__device__ __forceinline__ float4 fma4(float s, float4 a, float4 acc) {
    return make_float4(fmaf(s,a.x,acc.x), fmaf(s,a.y,acc.y), fmaf(s,a.z,acc.z), fmaf(s,a.w,acc.w));
}
__device__ __forceinline__ float4 add4(float4 a, float4 b) {
    return make_float4(a.x+b.x, a.y+b.y, a.z+b.z, a.w+b.w);
}
__device__ __forceinline__ float4 relu4(float4 a) {
    return make_float4(fmaxf(a.x,0.f), fmaxf(a.y,0.f), fmaxf(a.z,0.f), fmaxf(a.w,0.f));
}

// ---------------------------------------------------------------------------
// Generic bf16 WMMA GEMM:  out[M,N] = A[M,K] @ B[K,N] + bias[N]
// A is f32 or bf16 in global (converted to bf16 while staging to LDS).
// Bt is column-major bf16 [Nld_pad][K]  (Bt[n*K+k] = B[k][n]).
// Wave tile: 16M x (NT*16)N. Workgroup: MTILES x NGROUPS waves (256 threads).
// FUSE: out = resid + relu(acc + bias)   (fusion GEMM epilogue)
// ---------------------------------------------------------------------------
template <typename AT, int MT, int MTILES, int NGROUPS, int NT, int KDIM, bool FUSE>
__global__ __launch_bounds__(256) void gemm_wmma_bf16(
    const AT* __restrict__ A, const bf16_t* __restrict__ Bt,
    const float* __restrict__ bias, const float* __restrict__ resid,
    float* __restrict__ out, int Nld)
{
    constexpr int KP = KDIM + 8;            // pad rows to dodge LDS bank conflicts
    __shared__ bf16_t As[MT * KP];

    const int  tid  = threadIdx.x;
    const int  lane = tid & 31;
    const int  wv   = tid >> 5;
    const int  mi   = wv % MTILES;
    const int  gi   = wv / MTILES;
    const long mBase = (long)blockIdx.x * MT;

    // stage A tile (MT x KDIM) -> LDS as bf16, coalesced 8-elem chunks
    for (int idx = tid * 8; idx < MT * KDIM; idx += 256 * 8) {
        const int r = idx / KDIM, k = idx % KDIM;
        typename vec8<AT>::type t = *(const typename vec8<AT>::type*)&A[(mBase + r) * KDIM + k];
        v8bf o;
        #pragma unroll
        for (int c = 0; c < 8; ++c) o[c] = (bf16_t)t[c];
        *(v8bf*)&As[r * KP + k] = o;
    }
    __syncthreads();

    const int ln   = lane & 15;
    const int hi   = (lane >> 4) & 1;
    const int aRow = mi * 16 + ln;
    const int klo  = hi * 16;   // A frag: K runs [klo,klo+8) and [8+klo,8+klo+8)
    const int klo2 = hi * 16;   // B frag: K run  [klo2, klo2+16)

    v8f acc[NT];
    #pragma unroll
    for (int t = 0; t < NT; ++t)
        #pragma unroll
        for (int i = 0; i < 8; ++i) acc[t][i] = 0.f;

    // ---- software-pipelined K loop: prefetch step k0+32 while running k0 ----
    v8bf  a0 = *(const v8bf*)&As[aRow * KP + klo];
    v8bf  a1 = *(const v8bf*)&As[aRow * KP + 8 + klo];
    v16bf bcur[NT];
    #pragma unroll
    for (int t = 0; t < NT; ++t) {
        const int n = gi * NT * 16 + t * 16 + ln;
        bcur[t] = *(const v16bf*)&Bt[(long)n * KDIM + klo2];
    }

    #pragma unroll
    for (int k0 = 0; k0 < KDIM; k0 += 32) {
        v8bf  na0 = a0, na1 = a1;
        v16bf bnext[NT];
        if (k0 + 32 < KDIM) {
            na0 = *(const v8bf*)&As[aRow * KP + (k0 + 32) + klo];
            na1 = *(const v8bf*)&As[aRow * KP + (k0 + 32) + 8 + klo];
            #pragma unroll
            for (int t = 0; t < NT; ++t) {
                const int n = gi * NT * 16 + t * 16 + ln;
                bnext[t] = *(const v16bf*)&Bt[(long)n * KDIM + (k0 + 32) + klo2];
            }
        } else {
            #pragma unroll
            for (int t = 0; t < NT; ++t) bnext[t] = bcur[t];
        }

        // fence: keep the k0+32 load group issued ahead of the k0 WMMAs
        SCHED_FENCE();

        const v16bf av = __builtin_shufflevector(
            a0, a1, 0,1,2,3,4,5,6,7,8,9,10,11,12,13,14,15);
        #pragma unroll
        for (int t = 0; t < NT; ++t)
            acc[t] = __builtin_amdgcn_wmma_f32_16x16x32_bf16(
                false, av, false, bcur[t], (short)0, acc[t], false, false);

        // fence: keep this step's WMMAs from sliding into the next load group
        SCHED_FENCE();

        a0 = na0; a1 = na1;
        #pragma unroll
        for (int t = 0; t < NT; ++t) bcur[t] = bnext[t];
    }

    // ---- epilogue: bias (+ optional relu + residual), coalesced stores ----
    #pragma unroll
    for (int t = 0; t < NT; ++t) {
        const int   n  = gi * NT * 16 + t * 16 + ln;
        const float bn = bias[n];
        #pragma unroll
        for (int r = 0; r < 8; ++r) {
            const long row = mBase + mi * 16 + r + hi * 8;   // C layout: M = r + 8*hi
            float val = acc[t][r] + bn;
            if (FUSE) val = fmaxf(val, 0.f) + resid[row * Nld + n];
            out[row * Nld + n] = val;
        }
    }
}

// ---------------------------------------------------------------------------
// prep kernels
// ---------------------------------------------------------------------------
__global__ void zerobuf(float* p, int n) {
    for (int i = threadIdx.x; i < n; i += 256) p[i] = 0.f;
}

// Bt[n*K+k] = (n < N) ? W[k*N+n] : 0   (transpose + f32->bf16, zero-pad N)
__global__ void wtrans(const float* __restrict__ W, bf16_t* __restrict__ Bt,
                       int K, int N, int Npad)
{
    long idx = (long)blockIdx.x * 256 + threadIdx.x;
    if (idx >= (long)Npad * K) return;
    int n = (int)(idx / K), k = (int)(idx % K);
    Bt[idx] = (n < N) ? (bf16_t)W[(long)k * N + n] : (bf16_t)0.f;
}

__global__ void padbias(const float* __restrict__ ba, float* __restrict__ ba96) {
    int i = threadIdx.x;
    if (i < 96) ba96[i] = (i < 81) ? ba[i] : 0.f;
}

// ---------------------------------------------------------------------------
// softmax over q for each (pixel, p):  a_s[n*81 + p*9 + q]
// ---------------------------------------------------------------------------
__global__ __launch_bounds__(256) void softmax9(const float* __restrict__ alog,
                                                float* __restrict__ a_s)
{
    long t = (long)blockIdx.x * 256 + threadIdx.x;
    if (t >= (long)NPIX * 9) return;
    long row = t / 9; int p = (int)(t % 9);
    const float* src = alog + row * 96 + p * 9;
    float vv[9], m = -3.4e38f;
    #pragma unroll
    for (int q = 0; q < 9; ++q) { vv[q] = src[q]; m = fmaxf(m, vv[q]); }
    float s = 0.f;
    #pragma unroll
    for (int q = 0; q < 9; ++q) { vv[q] = __expf(vv[q] - m); s += vv[q]; }
    const float inv = 1.f / s;
    float* dst = a_s + row * 81 + p * 9;
    #pragma unroll
    for (int q = 0; q < 9; ++q) dst[q] = vv[q] * inv;
}

// ---------------------------------------------------------------------------
// Fused: outlook attention (unfold/einsum/fold collapsed to a per-pixel 5x5
// data-dependent conv over v) + relu + 3x3/5x5 maxpools of x + relus +
// concat -> cc (bf16, [NPIX,512]).  One wave per output pixel; lanes split C.
// Grid: 2 blocks per (b,s1) row for occupancy; uniform barrier counts.
// ---------------------------------------------------------------------------
__global__ __launch_bounds__(256) void attn_fold_pool(
    const float* __restrict__ v, const float* __restrict__ a_s,
    const float* __restrict__ x, bf16_t* __restrict__ cc)
{
    __shared__ float Wsh[8][32];
    const int tid = threadIdx.x, lane = tid & 31, wv = tid >> 5;
    const int bs = blockIdx.x >> 1;        // b*64 + s1
    const int half = blockIdx.x & 1;
    const int b = bs >> 6, s1 = bs & 63;

    for (int s2 = half * 32 + wv; s2 < half * 32 + 32; s2 += 8) {
        // phase 1: lanes 0..24 each build one of 25 conv weights
        if (lane < 25) {
            const int du = lane / 5 - 2;   // offset applied to s1
            const int dv = lane % 5 - 2;   // offset applied to s2
            float wsum = 0.f;
            #pragma unroll
            for (int i = 0; i < 3; ++i) {
                const int i2 = dv + i;
                const int hq = s2 + 1 - i;
                if (i2 < 0 || i2 > 2 || hq < 0 || hq > 63) continue;
                #pragma unroll
                for (int j = 0; j < 3; ++j) {
                    const int j2 = du + j;
                    const int wq = s1 + 1 - j;
                    if (j2 < 0 || j2 > 2 || wq < 0 || wq > 63) continue;
                    const long n = ((long)b * 64 + hq) * 64 + wq;
                    wsum += a_s[n * 81 + (i * 3 + j) * 9 + (i2 * 3 + j2)];
                }
            }
            Wsh[wv][lane] = wsum;
        }
        __syncthreads();

        // phase 2: 8 channels per lane, sweep the 5x5 neighborhood once,
        // sharing loads between the weighted v-sum and the two maxpools of x.
        const int c0 = lane * 8;
        float4 acc0 = {0,0,0,0}, acc1 = {0,0,0,0};
        float4 m1a0, m1a1, m2a0, m2a1;
        m1a0 = m1a1 = m2a0 = m2a1 = make_float4(-3.4e38f,-3.4e38f,-3.4e38f,-3.4e38f);

        #pragma unroll
        for (int du = -2; du <= 2; ++du) {
            const int h = s1 + du;
            if (h < 0 || h > 63) continue;
            #pragma unroll
            for (int dv = -2; dv <= 2; ++dv) {
                const int w2 = s2 + dv;
                if (w2 < 0 || w2 > 63) continue;
                const long base = (((long)b * 64 + h) * 64 + w2) * CDIM + c0;
                const float4 xv0 = ld4(x + base), xv1 = ld4(x + base + 4);
                m2a0 = max4(m2a0, xv0); m2a1 = max4(m2a1, xv1);
                if (du >= -1 && du <= 1 && dv >= -1 && dv <= 1) {
                    m1a0 = max4(m1a0, xv0); m1a1 = max4(m1a1, xv1);
                }
                const float wgt = Wsh[wv][(du + 2) * 5 + (dv + 2)];
                const float4 vv0 = ld4(v + base), vv1 = ld4(v + base + 4);
                acc0 = fma4(wgt, vv0, acc0); acc1 = fma4(wgt, vv1, acc1);
            }
        }

        // x1 = relu(relu(xf)+m1); x2 = relu(x1+m2)
        const float4 x1_0 = relu4(add4(relu4(acc0), m1a0));
        const float4 x1_1 = relu4(add4(relu4(acc1), m1a1));
        const float4 x2_0 = relu4(add4(x1_0, m2a0));
        const float4 x2_1 = relu4(add4(x1_1, m2a1));

        const long row = ((long)b * 64 + s1) * 64 + s2;
        v8bf p1, p2;
        p1[0]=(bf16_t)x1_0.x; p1[1]=(bf16_t)x1_0.y; p1[2]=(bf16_t)x1_0.z; p1[3]=(bf16_t)x1_0.w;
        p1[4]=(bf16_t)x1_1.x; p1[5]=(bf16_t)x1_1.y; p1[6]=(bf16_t)x1_1.z; p1[7]=(bf16_t)x1_1.w;
        p2[0]=(bf16_t)x2_0.x; p2[1]=(bf16_t)x2_0.y; p2[2]=(bf16_t)x2_0.z; p2[3]=(bf16_t)x2_0.w;
        p2[4]=(bf16_t)x2_1.x; p2[5]=(bf16_t)x2_1.y; p2[6]=(bf16_t)x2_1.z; p2[7]=(bf16_t)x2_1.w;
        *(v8bf*)&cc[row * 512 + c0]       = p1;
        *(v8bf*)&cc[row * 512 + 256 + c0] = p2;
    }
}

// ---------------------------------------------------------------------------
// BatchNorm stats: per-channel sum & sumsq via global f32 atomics
// ---------------------------------------------------------------------------
__global__ __launch_bounds__(256) void stats_kernel(const float* __restrict__ p,
                                                    float* __restrict__ stats)
{
    const int c = threadIdx.x;
    const long r0 = (long)blockIdx.x * 128;
    float s = 0.f, sq = 0.f;
    for (int r = 0; r < 128; ++r) {
        const float t = p[(r0 + r) * CDIM + c];
        s += t; sq += t * t;
    }
    atomicAdd(&stats[c], s);
    atomicAdd(&stats[CDIM + c], sq);
}

__global__ void bn_coef(const float* __restrict__ stats, const float* __restrict__ gamma,
                        const float* __restrict__ beta, float* __restrict__ coef)
{
    const int c = threadIdx.x;
    const float inv_n = 1.f / (float)NPIX;
    const float mean = stats[c] * inv_n;
    const float var  = stats[CDIM + c] * inv_n - mean * mean;
    const float sc   = gamma[c] * rsqrtf(var + EPSBN);
    coef[c]        = sc;
    coef[CDIM + c] = beta[c] - mean * sc;
}

// Out[b,i,j,c] = BN(pre)[b,j,i,c]  (spatial transpose; C innermost -> coalesced)
__global__ __launch_bounds__(256) void bn_transpose(const float* __restrict__ pre,
                                                    const float* __restrict__ coef,
                                                    float* __restrict__ out)
{
    const int tid = threadIdx.x;
    const long pix = (long)blockIdx.x * 4 + (tid >> 6);
    const int cg = (tid & 63) * 4;
    const int b  = (int)(pix >> 12);
    const int ij = (int)(pix & 4095);
    const int i  = ij >> 6, j = ij & 63;
    const long src = ((long)b * 64 + j) * 64 + i;
    const float4 v4 = ld4(pre + src * CDIM + cg);
    const float4 s4 = ld4(coef + cg);
    const float4 h4 = ld4(coef + CDIM + cg);
    float4 o;
    o.x = fmaf(v4.x, s4.x, h4.x); o.y = fmaf(v4.y, s4.y, h4.y);
    o.z = fmaf(v4.z, s4.z, h4.z); o.w = fmaf(v4.w, s4.w, h4.w);
    *(float4*)&out[pix * CDIM + cg] = o;
}

// ---------------------------------------------------------------------------
// launch
// ---------------------------------------------------------------------------
extern "C" void kernel_launch(void* const* d_in, const int* in_sizes, int n_in,
                              void* d_out, int out_size, void* d_ws, size_t ws_size,
                              hipStream_t stream)
{
    const float* x     = (const float*)d_in[0];   // [8,64,64,256]
    const float* Wv    = (const float*)d_in[1];   // [256,256]
    const float* bv    = (const float*)d_in[2];   // [256]
    const float* Wa    = (const float*)d_in[3];   // [256,81]
    const float* ba    = (const float*)d_in[4];   // [81]
    const float* Wfu   = (const float*)d_in[5];   // [512,256]
    const float* bfu   = (const float*)d_in[6];   // [256]
    const float* gamma = (const float*)d_in[7];   // [256]
    const float* beta  = (const float*)d_in[8];   // [256]
    float* out = (float*)d_out;

    char* ws = (char*)d_ws;
    // workspace layout (~86.6 MiB total); v is dead before the fusion GEMM
    // writes out_pre, so they alias.
    float*  v_buf   = (float*) (ws + 0);            // 33,554,432 B  [NPIX,256] f32
    float*  out_pre = v_buf;                        // alias (v dead after attn)
    float*  alog    = (float*) (ws + 33554432);     // 12,582,912 B  [NPIX,96]  f32
    float*  a_s     = (float*) (ws + 46137344);     // 10,616,832 B  [NPIX,81]  f32
    bf16_t* cc      = (bf16_t*)(ws + 56754176);     // 33,554,432 B  [NPIX,512] bf16
    bf16_t* WvT     = (bf16_t*)(ws + 90308608);     //    131,072 B  [256][256]
    bf16_t* WaT     = (bf16_t*)(ws + 90439680);     //     49,152 B  [96][256]
    bf16_t* WfuT    = (bf16_t*)(ws + 90488832);     //    262,144 B  [256][512]
    float*  ba96    = (float*) (ws + 90750976);     //        512 B
    float*  stats   = (float*) (ws + 90751488);     //      2,048 B
    float*  coef    = (float*) (ws + 90753536);     //      2,048 B

    zerobuf<<<1, 256, 0, stream>>>(stats, 512);
    wtrans<<<256, 256, 0, stream>>>(Wv, WvT, 256, 256, 256);
    wtrans<<<96, 256, 0, stream>>>(Wa, WaT, 256, 81, 96);
    wtrans<<<512, 256, 0, stream>>>(Wfu, WfuT, 512, 256, 256);
    padbias<<<1, 128, 0, stream>>>(ba, ba96);

    // v = x @ Wv + bv                       (M=32768,K=256,N=256)
    gemm_wmma_bf16<float, 32, 2, 4, 4, 256, false>
        <<<NPIX / 32, 256, 0, stream>>>(x, WvT, bv, nullptr, v_buf, 256);
    // attention logits = x @ Wa + ba (padded to 96 cols)
    gemm_wmma_bf16<float, 64, 4, 2, 3, 256, false>
        <<<NPIX / 64, 256, 0, stream>>>(x, WaT, ba96, nullptr, alog, 96);

    softmax9<<<(NPIX * 9) / 256, 256, 0, stream>>>(alog, a_s);

    attn_fold_pool<<<BB * HWDIM * 2, 256, 0, stream>>>(v_buf, a_s, x, cc);

    // out_pre = x + relu(cc @ Wfu + bfu)    (M=32768,K=512,N=256)
    gemm_wmma_bf16<bf16_t, 32, 2, 4, 4, 512, true>
        <<<NPIX / 32, 256, 0, stream>>>(cc, WfuT, bfu, x, out_pre, 256);

    stats_kernel<<<NPIX / 128, 256, 0, stream>>>(out_pre, stats);
    bn_coef<<<1, 256, 0, stream>>>(stats, gamma, beta, coef);
    bn_transpose<<<NPIX / 4, 256, 0, stream>>>(out_pre, coef, out);
}
